// GraphTRMv2_51135880626830
// MI455X (gfx1250) — matrix operand order repaired and checked by reading, not compiled, not run on hardware
//
#include <hip/hip_runtime.h>
#include <stdint.h>

// ---------------------------------------------------------------------------
// Types for CDNA5 WMMA (wave32, v_wmma_f32_16x16x32_bf16)
// ---------------------------------------------------------------------------
typedef __attribute__((ext_vector_type(16))) __bf16 v16bf;
typedef __attribute__((ext_vector_type(8)))  __bf16 v8bf;
typedef __attribute__((ext_vector_type(8)))  float  v8f;

#define WG 256   // 8 wave32 per block

static constexpr int NN = 20000;
static constexpr int NE = 320000;
static constexpr int HD = 256;

// ---------------------------------------------------------------------------
// device helpers
// ---------------------------------------------------------------------------
__device__ __forceinline__ float gelu_exact(float v) {
  return 0.5f * v * (1.0f + erff(v * 0.70710678118654752f));
}
__device__ __forceinline__ float sigm(float v) { return 1.0f / (1.0f + __expf(-v)); }
__device__ __forceinline__ float logsig(float x) {
  return x >= 0.f ? -log1pf(__expf(-x)) : x - log1pf(__expf(x));
}
__device__ __forceinline__ void atomAddF(float* p, float v) { unsafeAtomicAdd(p, v); }

// A fragment: 16x32 bf16 tile, row-major source with leading dim lda.
// Layout: lanes 0-15 row M=lane hold K 0..15; lanes 16-31 same rows,
// K 16..31 (matches load_b).  => two contiguous 16B loads.
__device__ __forceinline__ v16bf load_a(const __bf16* A, int lda, int row0, int k0, int lane) {
  int m = lane & 15, kh = lane >> 4;
  const __bf16* p = A + (size_t)(row0 + m) * lda + k0 + kh * 16;
  v8bf lo = *(const v8bf*)p;
  v8bf hi = *(const v8bf*)(p + 8);
  v16bf r;
#pragma unroll
  for (int i = 0; i < 8; i++) { r[i] = lo[i]; r[i + 8] = hi[i]; }
  return r;
}

// B fragment: 32x16 bf16 tile from pre-transposed weights Wt[NOUT][K]
// (column n of the GEMM is a contiguous K-row).  Layout (same as load_a): lanes 0-15
// col N=lane hold K=k0..k0+15; lanes 16-31 hold K=k0+16..k0+31.
__device__ __forceinline__ v16bf load_b(const __bf16* Wt, int K, int col0, int k0, int lane) {
  int n = lane & 15, ks = lane >> 4;
  const __bf16* p = Wt + (size_t)(col0 + n) * K + k0 + ks * 16;
  return *(const v16bf*)p;
}

// ---------------------------------------------------------------------------
// GEMM + fused epilogue.  C[16 x NOUT] per block, 8 waves, each wave owns
// NOUT/128 column tiles.  Epilogue modes:
//  0: +bias +sigmoid(y[row])*r1[col]; LayerNorm(gamma,beta)        (LP)
//  1: +bias +y[row]*r1[col];          LayerNorm(gamma,beta)        (OP)
//  2: +bias; LayerNorm; GELU                                       (GIN mlp1)
//  3: GELU(+bias); +residual; LayerNorm                            (GIN mlp2)
//  4: +bias; GELU                                                  (head mlp1)
// ---------------------------------------------------------------------------
template <int NOUT>
__global__ __launch_bounds__(WG) void gemm_epi(
    const __bf16* A, int lda, int K,
    const __bf16* __restrict__ Wt,
    const float* __restrict__ bias,
    const float* __restrict__ gamma, const float* __restrict__ beta,
    const float* __restrict__ yvec, const float* __restrict__ r1,
    const __bf16* resid,
    __bf16* out, int ldo,
    __bf16* out2, int ldo2,
    int mode) {
  constexpr int TILES = NOUT / (16 * 8);
  __shared__ float smem[16 * NOUT];
  __shared__ float srow[16];

  const int tid = threadIdx.x;
  const int lane = tid & 31, wave = tid >> 5;
  const int row0 = blockIdx.x * 16;
  const int colbase = wave * (NOUT / 8);

  v8f acc[TILES];
#pragma unroll
  for (int t = 0; t < TILES; t++)
#pragma unroll
    for (int r = 0; r < 8; r++) acc[t][r] = 0.0f;

  for (int k0 = 0; k0 < K; k0 += 32) {
    v16bf a = load_a(A, lda, row0, k0, lane);
    if (k0 + 32 < K)  // gfx1250 global_prefetch_b8 for next A strip
      __builtin_prefetch(A + (size_t)(row0 + (lane & 15)) * lda + k0 + 32, 0, 3);
#pragma unroll
    for (int t = 0; t < TILES; t++) {
      v16bf b = load_b(Wt, K, colbase + t * 16, k0, lane);
      acc[t] = __builtin_amdgcn_wmma_f32_16x16x32_bf16(
          false, a, false, b, (short)0, acc[t], false, false);
    }
  }

  // per-row rank-1 scalar
  if (tid < 16) {
    float s = 0.f;
    if (mode == 0) s = sigm(yvec[row0 + tid]);
    else if (mode == 1) s = yvec[row0 + tid];
    srow[tid] = s;
  }
  __syncthreads();

  // scatter accumulators to LDS, fusing bias / rank-1 / (gelu+residual)
  const int nlo = lane & 15, mhi = (lane >> 4) * 8;
#pragma unroll
  for (int t = 0; t < TILES; t++) {
    int col = colbase + t * 16 + nlo;
    float bcol = bias[col];
    float r1c = (mode <= 1) ? r1[col] : 0.f;
#pragma unroll
    for (int r = 0; r < 8; r++) {
      int row = mhi + r;
      float v = acc[t][r] + bcol;
      if (mode == 3)
        v = gelu_exact(v) + (float)resid[(size_t)(row0 + row) * 256 + col];
      else
        v += srow[row] * r1c;
      smem[row * NOUT + col] = v;
    }
  }
  __syncthreads();

  // row LayerNorm (16 threads per row; shfl reduce within 16-lane groups)
  const int r = tid >> 4, j = tid & 15;
  float mean = 0.f, rstd = 0.f;
  const bool do_ln = (mode != 4);
  if (do_ln) {
    float s = 0.f, q = 0.f;
    for (int c = j; c < NOUT; c += 16) {
      float v = smem[r * NOUT + c];
      s += v; q += v * v;
    }
#pragma unroll
    for (int off = 8; off >= 1; off >>= 1) {
      s += __shfl_xor(s, off, 16);
      q += __shfl_xor(q, off, 16);
    }
    mean = s / (float)NOUT;
    float var = q / (float)NOUT - mean * mean;
    rstd = rsqrtf(var + 1e-5f);
  }
  for (int c = j; c < NOUT; c += 16) {
    float v = smem[r * NOUT + c];
    if (do_ln) v = (v - mean) * rstd * gamma[c] + beta[c];
    if (mode == 2 || mode == 4) v = gelu_exact(v);
    __bf16 o = (__bf16)v;
    out[(size_t)(row0 + r) * ldo + c] = o;
    if (out2) out2[(size_t)(row0 + r) * ldo2 + c] = o;
  }
}

// ---------------------------------------------------------------------------
// small kernels
// ---------------------------------------------------------------------------
__global__ void zero_kernel(uint32_t* p, size_t nwords) {
  size_t i = (size_t)blockIdx.x * blockDim.x + threadIdx.x;
  size_t stride = (size_t)gridDim.x * blockDim.x;
  for (; i < nwords; i += stride) p[i] = 0u;
}

__global__ void pack_kernel(const float* __restrict__ src, int srcLd, int rowStart,
                            __bf16* __restrict__ dst, int dstStride, int dstOff,
                            int K, int NO) {
  int idx = blockIdx.x * WG + threadIdx.x;
  if (idx >= K * NO) return;
  int n = idx / K, k = idx % K;
  dst[(size_t)n * dstStride + dstOff + k] = (__bf16)src[(size_t)(rowStart + k) * srcLd + n];
}

__global__ void deg_kernel(const int* __restrict__ ei, float* __restrict__ deg, int E) {
  int e = blockIdx.x * WG + threadIdx.x;
  if (e < E) atomAddF(&deg[ei[e]], 1.0f);
}

__global__ void yinit_kernel(const float* __restrict__ deg, float* __restrict__ y, int N) {
  int i = blockIdx.x * WG + threadIdx.x;
  if (i >= N) return;
  float p = 1.0f / (1.0f + deg[i]);
  p = fminf(fmaxf(p, 0.01f), 0.99f);
  y[i] = logf(p / (1.0f - p));
}

__global__ __launch_bounds__(WG) void xemb_kernel(
    const float* __restrict__ x, const float* __restrict__ xw,
    const float* __restrict__ xb, const float* __restrict__ g,
    const float* __restrict__ b, __bf16* __restrict__ act) {
  int i = blockIdx.x, c = threadIdx.x;
  float v = x[2 * i] * xw[c] + x[2 * i + 1] * xw[256 + c] + xb[c];
  __shared__ float rs[8], rq[8];
  float s = v, q = v * v;
#pragma unroll
  for (int off = 16; off >= 1; off >>= 1) {
    s += __shfl_xor(s, off, 32);
    q += __shfl_xor(q, off, 32);
  }
  int lane = c & 31, w = c >> 5;
  if (lane == 0) { rs[w] = s; rq[w] = q; }
  __syncthreads();
  s = 0.f; q = 0.f;
#pragma unroll
  for (int k = 0; k < 8; k++) { s += rs[k]; q += rq[k]; }
  float m = s / 256.f;
  float var = q / 256.f - m * m;
  act[(size_t)i * 512 + c] = (__bf16)((v - m) * rsqrtf(var + 1e-5f) * g[c] + b[c]);
}

// agg = (1+eps)*h  (fp32)
__global__ void scale_init_kernel(const __bf16* __restrict__ h,
                                  const float* __restrict__ epsp,
                                  float* __restrict__ agg) {
  size_t idx = (size_t)blockIdx.x * WG + threadIdx.x;
  agg[idx] = (1.0f + epsp[0]) * (float)h[idx];
}

// agg[dst] += h[src] over edges; hardware global_atomic_add_f32, L2-resident
__global__ __launch_bounds__(WG) void scatter_kernel(
    const int* __restrict__ ei, const __bf16* __restrict__ h,
    float* __restrict__ agg, int E) {
  int c = threadIdx.x;
  int ebase = blockIdx.x * 8;
#pragma unroll
  for (int j = 0; j < 8; j++) {
    int e = ebase + j;
    if (e >= E) break;
    int s = ei[e], d = ei[E + e];
    float v = (float)h[(size_t)s * HD + c];
    atomAddF(&agg[(size_t)d * HD + c], v);
  }
}

__global__ void convert_kernel(const float* __restrict__ agg, __bf16* __restrict__ m) {
  size_t idx = (size_t)blockIdx.x * WG + threadIdx.x;
  m[idx] = (__bf16)agg[idx];
}

// y = g @ hw2 + hb2   (one wave per node)
__global__ __launch_bounds__(WG) void yupdate_kernel(
    const __bf16* __restrict__ gbuf, const float* __restrict__ hw2,
    const float* __restrict__ hb2, float* __restrict__ y, int N) {
  int wave = threadIdx.x >> 5, lane = threadIdx.x & 31;
  int i = blockIdx.x * 8 + wave;
  if (i >= N) return;
  float s = 0.f;
  for (int c = lane; c < HD; c += 32) s += (float)gbuf[(size_t)i * HD + c] * hw2[c];
#pragma unroll
  for (int off = 16; off >= 1; off >>= 1) s += __shfl_xor(s, off, 32);
  if (lane == 0) y[i] = s + hb2[0];
}

__device__ __forceinline__ float block_sum_256(float v) {
#pragma unroll
  for (int off = 16; off >= 1; off >>= 1) v += __shfl_xor(v, off, 32);
  __shared__ float r[8];
  int lane = threadIdx.x & 31, w = threadIdx.x >> 5;
  if (lane == 0) r[w] = v;
  __syncthreads();
  float s = 0.f;
  if (w == 0) {
    s = (lane < 8) ? r[lane] : 0.f;
#pragma unroll
    for (int off = 4; off >= 1; off >>= 1) s += __shfl_xor(s, off, 8);
  }
  return s;  // valid on thread 0
}

__global__ __launch_bounds__(WG) void pos_kernel(const int* __restrict__ labels,
                                                 float* scal, int N) {
  int i = blockIdx.x * WG + threadIdx.x;
  float v = (i < N) ? (float)labels[i] : 0.f;
  float s = block_sum_256(v);
  if (threadIdx.x == 0) atomAddF(&scal[0], s);
}

__global__ __launch_bounds__(WG) void bce_kernel(const float* __restrict__ y,
                                                 const int* __restrict__ labels,
                                                 float* scal, int N) {
  int i = blockIdx.x * WG + threadIdx.x;
  float pos = fmaxf(scal[0], 1.f);
  float neg = fmaxf((float)N - pos, 1.f);
  float pw = neg / pos;
  float v = 0.f;
  if (i < N) {
    float lg = fminf(fmaxf(y[i], -10.f), 10.f);
    float lab = (float)labels[i];
    v = -(pw * lab * logsig(lg) + (1.f - lab) * logsig(-lg));
  }
  float s = block_sum_256(v);
  if (threadIdx.x == 0) atomAddF(&scal[1], s);
}

__global__ __launch_bounds__(WG) void feas_kernel(const float* __restrict__ y,
                                                  const int* __restrict__ ei,
                                                  float* scal, int E) {
  int e = blockIdx.x * WG + threadIdx.x;
  float v = 0.f;
  if (e < E) {
    float ls = fminf(fmaxf(y[ei[e]], -10.f), 10.f);
    float ld = fminf(fmaxf(y[ei[E + e]], -10.f), 10.f);
    v = sigm(ls) * sigm(ld);
  }
  float s = block_sum_256(v);
  if (threadIdx.x == 0) atomAddF(&scal[2], s);
}

__global__ void final_kernel(const float* __restrict__ scal, float* __restrict__ out) {
  if (threadIdx.x == 0 && blockIdx.x == 0)
    out[0] = scal[1] / (float)NN + 50.0f * scal[2] / (float)NE;
}

// ---------------------------------------------------------------------------
// host
// ---------------------------------------------------------------------------
extern "C" void kernel_launch(void* const* d_in, const int* in_sizes, int n_in,
                              void* d_out, int out_size, void* d_ws, size_t ws_size,
                              hipStream_t stream) {
  (void)in_sizes; (void)n_in; (void)out_size; (void)ws_size;
  const int N = NN, E = NE;

  // inputs in setup_inputs() dict order (params flattened in insertion order)
  const float* x      = (const float*)d_in[0];
  const int*   ei     = (const int*)d_in[1];
  const int*   labels = (const int*)d_in[2];
  const float* xw  = (const float*)d_in[3];
  const float* xb  = (const float*)d_in[4];
  const float* xng = (const float*)d_in[5];
  const float* xnb = (const float*)d_in[6];
  const float* lpw = (const float*)d_in[7];
  const float* lpb = (const float*)d_in[8];
  const float* lng = (const float*)d_in[9];
  const float* lnb = (const float*)d_in[10];
  struct Gin { const float *eps,*w1,*b1,*g1,*be1,*w2,*b2,*ng,*nb; } gin[2];
  for (int l = 0; l < 2; l++) {
    int b0 = 11 + l * 9;
    gin[l].eps = (const float*)d_in[b0 + 0];
    gin[l].w1  = (const float*)d_in[b0 + 1];
    gin[l].b1  = (const float*)d_in[b0 + 2];
    gin[l].g1  = (const float*)d_in[b0 + 3];
    gin[l].be1 = (const float*)d_in[b0 + 4];
    gin[l].w2  = (const float*)d_in[b0 + 5];
    gin[l].b2  = (const float*)d_in[b0 + 6];
    gin[l].ng  = (const float*)d_in[b0 + 7];
    gin[l].nb  = (const float*)d_in[b0 + 8];
  }
  const float* opw = (const float*)d_in[29];
  const float* opb = (const float*)d_in[30];
  const float* ong = (const float*)d_in[31];
  const float* onb = (const float*)d_in[32];
  const float* hw1 = (const float*)d_in[33];
  const float* hb1 = (const float*)d_in[34];
  const float* hw2 = (const float*)d_in[35];
  const float* hb2 = (const float*)d_in[36];

  // workspace carve-out
  char* ws = (char*)d_ws;
  size_t off = 0;
  auto carve = [&](size_t bytes) -> void* {
    void* p = ws + off;
    off = (off + bytes + 255) & ~(size_t)255;
    return p;
  };
  __bf16* act  = (__bf16*)carve((size_t)N * 512 * 2);  // [x_emb | z]
  __bf16* h    = (__bf16*)carve((size_t)N * 256 * 2);
  __bf16* m_in = (__bf16*)carve((size_t)N * 256 * 2);
  void*   R1   = carve((size_t)N * 512 * 2);           // aggf(f32,256) / m(bf16,512) / ho|g
  float*  aggf = (float*)R1;
  __bf16* mbuf = (__bf16*)R1;
  __bf16* ho   = (__bf16*)R1;
  __bf16* gbuf = (__bf16*)R1 + (size_t)N * 256;
  float*  y    = (float*)carve((size_t)N * 4);
  float*  deg  = (float*)carve((size_t)N * 4);
  __bf16* lpwT = (__bf16*)carve((size_t)256 * 512 * 2);
  __bf16* w1T[2], *w2T[2];
  for (int l = 0; l < 2; l++) {
    w1T[l] = (__bf16*)carve((size_t)512 * 256 * 2);
    w2T[l] = (__bf16*)carve((size_t)256 * 512 * 2);
  }
  __bf16* opwT = (__bf16*)carve((size_t)256 * 256 * 2);
  __bf16* hw1T = (__bf16*)carve((size_t)256 * 256 * 2);
  float*  scal = (float*)carve(64);

  // init: zero act (z half), deg, loss accumulators
  zero_kernel<<<2048, WG, 0, stream>>>((uint32_t*)act, (size_t)N * 512 * 2 / 4);
  zero_kernel<<<80, WG, 0, stream>>>((uint32_t*)deg, (size_t)N);
  zero_kernel<<<1, WG, 0, stream>>>((uint32_t*)scal, 16);

  // pack weights -> bf16, transposed to [NOUT][K] (contiguous K per column)
  auto pack = [&](const float* src, int srcLd, int rowStart, __bf16* dst,
                  int dstStride, int dstOff, int K, int NO) {
    pack_kernel<<<(K * NO + WG - 1) / WG, WG, 0, stream>>>(
        src, srcLd, rowStart, dst, dstStride, dstOff, K, NO);
  };
  pack(lpw, 256, 0,   lpwT, 512, 0,   256, 256);  // lpw rows 0..255
  pack(lpw, 256, 257, lpwT, 512, 256, 256, 256);  // lpw rows 257..512 (row 256 = rank-1)
  for (int l = 0; l < 2; l++) {
    pack(gin[l].w1, 512, 0, w1T[l], 256, 0, 256, 512);
    pack(gin[l].w2, 256, 0, w2T[l], 512, 0, 512, 256);
  }
  pack(opw, 256, 1, opwT, 256, 0, 256, 256);      // opw row 0 = rank-1
  pack(hw1, 256, 0, hw1T, 256, 0, 256, 256);

  // degree -> y init; x embedding into act[:, :256]
  deg_kernel<<<(E + WG - 1) / WG, WG, 0, stream>>>(ei, deg, E);
  yinit_kernel<<<(N + WG - 1) / WG, WG, 0, stream>>>(deg, y, N);
  xemb_kernel<<<N, WG, 0, stream>>>(x, xw, xb, xng, xnb, act);

  const int GBLK = N / 16;  // 1250
  auto g256 = [&](const __bf16* A, int lda, int K, const __bf16* Wt,
                  const float* bias, const float* gam, const float* bet,
                  const float* yv, const float* r1, const __bf16* res,
                  __bf16* out, int ldo, __bf16* out2, int ldo2, int mode) {
    gemm_epi<256><<<GBLK, WG, 0, stream>>>(A, lda, K, Wt, bias, gam, bet, yv, r1,
                                           res, out, ldo, out2, ldo2, mode);
  };
  auto g512 = [&](const __bf16* A, int lda, int K, const __bf16* Wt,
                  const float* bias, const float* gam, const float* bet,
                  __bf16* out, int ldo, int mode) {
    gemm_epi<512><<<GBLK, WG, 0, stream>>>(A, lda, K, Wt, bias, gam, bet, nullptr,
                                           nullptr, nullptr, out, ldo, nullptr, 0, mode);
  };

  for (int hc = 0; hc < 3; hc++) {
    for (int lc = 0; lc < 6; lc++) {
      // h = LN([x_emb, sigmoid(y), z] @ lpw + lpb)  (rank-1 fused)
      g256(act, 512, 512, lpwT, lpb, lng, lnb, y, lpw + 256 * 256,
           nullptr, h, 256, nullptr, 0, /*mode*/0);
      for (int l = 0; l < 2; l++) {
        // agg = (1+eps)*h ; agg[dst] += h[src]
        scale_init_kernel<<<N * HD / WG, WG, 0, stream>>>(h, gin[l].eps, aggf);
        scatter_kernel<<<E / 8, WG, 0, stream>>>(ei, h, aggf, E);
        convert_kernel<<<N * HD / WG, WG, 0, stream>>>(aggf, m_in);
        // m = gelu(LN(m_in @ w1 + b1))
        g512(m_in, 256, 256, w1T[l], gin[l].b1, gin[l].g1, gin[l].be1,
             mbuf, 512, /*mode*/2);
        // h = LN(h + gelu(m @ w2 + b2)); also mirror into act z-half
        g256(mbuf, 512, 512, w2T[l], gin[l].b2, gin[l].ng, gin[l].nb,
             nullptr, nullptr, h, h, 256, act + 256, 512, /*mode*/3);
      }
    }
    // ho = LN([y, z] @ opw + opb)  (rank-1 fused)
    g256(act + 256, 512, 256, opwT, opb, ong, onb, y, opw,
         nullptr, ho, 256, nullptr, 0, /*mode*/1);
    // g = gelu(ho @ hw1 + hb1)
    g256(ho, 256, 256, hw1T, hb1, nullptr, nullptr, nullptr, nullptr,
         nullptr, gbuf, 256, nullptr, 0, /*mode*/4);
    // y = g @ hw2 + hb2
    yupdate_kernel<<<N / 8, WG, 0, stream>>>(gbuf, hw2, hb2, y, N);
  }

  // loss
  pos_kernel<<<(N + WG - 1) / WG, WG, 0, stream>>>(labels, scal, N);
  bce_kernel<<<(N + WG - 1) / WG, WG, 0, stream>>>(y, labels, scal, N);
  feas_kernel<<<E / WG, WG, 0, stream>>>(y, ei, scal, E);
  final_kernel<<<1, WG, 0, stream>>>(scal, (float*)d_out);
}
